// XLSTMRegressor_14748917694571
// MI455X (gfx1250) — matrix operand order repaired and checked by reading, not compiled, not run on hardware
//
#include <hip/hip_runtime.h>
#include <hip/hip_bf16.h>
#include <math.h>

// ---------------------------------------------------------------------------
// Problem constants (reference: B=128, T=2048, D_IN=32, H=256)
// ---------------------------------------------------------------------------
#define BB   128
#define TT   2048
#define DIN  32
#define HH   256
#define G4   1024          // 4*H
#define NWG  8             // batch tiles of 16 rows -> 8 workgroups for the scan
#define NCHUNK 16          // EMA chunks (T/128)

typedef __attribute__((ext_vector_type(16))) __bf16 v16bf;
typedef __attribute__((ext_vector_type(8)))  float  v8f;

// workspace byte offsets
#define OFF_WCOMB_F32   0u                       // 1024*32 f32      = 131072
#define OFF_BCOMB       131072u                  // 1024 f32         = 4096
#define OFF_WHH_SWZ     135168u                  // 1024*256 bf16    = 524288
#define OFF_WCOMB_SWZ   659456u                  // 1024*32 bf16     = 65536
#define OFF_HLAST       724992u                  // 128*256 f32      = 131072
#define OFF_EMALAST     856064u                  // 128*256 f32     = 131072
#define OFF_EMA_A       987136u                  // 128*16*256 f32   = 2097152
#define OFF_EMA_B       3084288u                 // 128*16*256 f32   = 2097152
#define OFF_XSWZ        5181440u                 // 8*2048*32*16 bf16= 16777216
// total ~21.9 MB

// ---------------------------------------------------------------------------
// helpers
// ---------------------------------------------------------------------------
union U32B { int4 q[2]; v16bf v; };

__device__ __forceinline__ v16bf ld32B(const __bf16* p) {
    U32B u;
    u.q[0] = ((const int4*)p)[0];
    u.q[1] = ((const int4*)p)[1];
    return u.v;
}

__device__ __forceinline__ v8f wmma_bf16(v16bf a, v16bf b, v8f c) {
    // D = A(16x32 bf16) * B(32x16 bf16) + C(16x16 f32)
    return __builtin_amdgcn_wmma_f32_16x16x32_bf16(false, a, false, b,
                                                   (short)0, c, false, false);
}

__device__ __forceinline__ float sigm(float x) {
    return 1.0f / (1.0f + __expf(-x));
}

__device__ __forceinline__ float ftanh(float x) {
    x = fminf(fmaxf(x, -15.f), 15.f);         // avoid inf/inf
    float e = __expf(2.f * x);
    return (e - 1.f) / (e + 1.f);
}

__device__ __forceinline__ unsigned short bf16bits(float f) {
    __bf16 b = (__bf16)f;
    unsigned short s;
    __builtin_memcpy(&s, &b, 2);
    return s;
}

// unpack element e (0/1) of a packed bf16 pair into f32
__device__ __forceinline__ float unpk(unsigned u, int e) {
    return __uint_as_float(e ? (u & 0xffff0000u) : (u << 16));
}

// permuted gate column p -> original gate row: p = 4*ch + gt, orig = gt*256 + ch
__device__ __forceinline__ int perm_orig(int p) {
    return (p & 3) * HH + (p >> 2);
}

// A-matrix (16-bit, 16x32) K position for (lane, vgpr dword v, half)
__device__ __forceinline__ int a_kpos(int lane, int v, int half) {
    return ((lane >= 16) ? 16 : 0) + 2 * v + half;
}

// ---- CDNA5 async global->LDS DMA (ASYNCcnt path), GV addressing mode ------
// 16 bytes per enabled lane, copied from global vaddr to LDS address vdst
__device__ __forceinline__ void async_g2l_b128(unsigned lds, unsigned long long gaddr) {
    asm volatile("global_load_async_to_lds_b128 %0, %1, off"
                 :: "v"(lds), "v"(gaddr) : "memory");
}
__device__ __forceinline__ void wait_async0() {
    asm volatile("s_wait_asynccnt 0x0" ::: "memory");
}

// ---------------------------------------------------------------------------
// Kernel 1: W_comb = W_ih @ W_in ; b_comb = W_ih @ b_in + b_ih + b_hh
// grid 1024 x 32
// ---------------------------------------------------------------------------
__global__ void prep_wcomb(const float* __restrict__ W_ih,   // [1024,256]
                           const float* __restrict__ W_in,   // [256,32]
                           const float* __restrict__ b_in,   // [256]
                           const float* __restrict__ b_ih,   // [1024]
                           const float* __restrict__ b_hh,   // [1024]
                           float* __restrict__ wcomb,        // [1024,32]
                           float* __restrict__ bcomb)        // [1024]
{
    int n = blockIdx.x, d = threadIdx.x;
    float acc = 0.f;
    for (int k = 0; k < HH; ++k)
        acc += W_ih[n * HH + k] * W_in[k * DIN + d];
    wcomb[n * DIN + d] = acc;
    if (d == 0) {
        float s = b_ih[n] + b_hh[n];
        for (int k = 0; k < HH; ++k) s += W_ih[n * HH + k] * b_in[k];
        bcomb[n] = s;
    }
}

// ---------------------------------------------------------------------------
// Kernel 2: swizzle W_hh into WMMA *A-matrix* layout (bf16), gate-interleaved
// permuted columns.  A[m][k] = W_hh[orig(perm_col)][k].
// grid 512 x 256
// ---------------------------------------------------------------------------
__global__ void pack_whh(const float* __restrict__ W_hh,  // [1024,256]
                         __bf16* __restrict__ swz)
{
    int id   = blockIdx.x * 256 + threadIdx.x;     // dword id, 0..131071
    int tile = id >> 8;                            // mt*8 + kc
    int r    = id & 255;
    int lane = r >> 3;
    int v    = r & 7;
    int mt = tile >> 3, kc = tile & 7;
    int p  = mt * 16 + (lane & 15);                // permuted column
    int n  = perm_orig(p);                         // original W_hh row
    int K  = kc * 32 + a_kpos(lane, v, 0);
    __bf16* dst = swz + ((size_t)(tile * 32 + lane) * 16 + 2 * v);
    dst[0] = (__bf16)W_hh[n * HH + K];
    dst[1] = (__bf16)W_hh[n * HH + K + 1];
}

// ---------------------------------------------------------------------------
// Kernel 3: swizzle W_comb into A-matrix layout (K=32, one chunk). grid 64x256
// ---------------------------------------------------------------------------
__global__ void pack_wcomb(const float* __restrict__ wcomb,  // [1024,32]
                           __bf16* __restrict__ swz)
{
    int id   = blockIdx.x * 256 + threadIdx.x;     // 0..16383
    int mt   = id >> 8;
    int r    = id & 255;
    int lane = r >> 3;
    int v    = r & 7;
    int p = mt * 16 + (lane & 15);
    int n = perm_orig(p);
    int K = a_kpos(lane, v, 0);
    __bf16* dst = swz + ((size_t)(mt * 32 + lane) * 16 + 2 * v);
    dst[0] = (__bf16)wcomb[n * DIN + K];
    dst[1] = (__bf16)wcomb[n * DIN + K + 1];
}

// ---------------------------------------------------------------------------
// Kernel 4: swizzle x into per-timestep WMMA *B-matrix* tiles (bf16).
// B[k][n] = x[row n][k]: lane l holds row l&15, K = (l>>4)*16 .. +15
// (contiguous). grid 2048 x 256
// ---------------------------------------------------------------------------
__global__ void pack_x(const float* __restrict__ x,   // [128,2048,32]
                       __bf16* __restrict__ xswz)
{
    int id   = blockIdx.x * 256 + threadIdx.x;   // (wg*2048 + t)*32 + lane
    int lane = id & 31;
    int t    = (id >> 5) & 2047;
    int wg   = id >> 16;
    int b    = wg * 16 + (lane & 15);
    int k0   = (lane >> 4) * 16;
    const float* src = x + ((size_t)b * TT + t) * DIN + k0;
    __bf16* dst = xswz + (size_t)id * 16;
    #pragma unroll
    for (int e = 0; e < 16; ++e) dst[e] = (__bf16)src[e];
}

// ---------------------------------------------------------------------------
// Kernel 5: sequential LSTM scan. 8 workgroups x 512 threads (16 waves).
// D = W(A) x h^T(B): lane = 1 batch row x 8 consecutive permuted gate columns
// per tile = i,f,g,o of 2 channels -> lane-local cell update, no gate
// exchange, one barrier/step (double-buffered h_bf16).
// W_hh kc0..3 register-resident, kc4..7 LDS-resident (async-DMA staged).
// x_t tile async-prefetched into double-buffered LDS staging by wave 0.
// x-part WMMA uses inline-0 C; biases added at use from packed bf16 pairs.
// ---------------------------------------------------------------------------
#define HBS 264   // h_bf16 row stride (528B = 132 dwords = 4 mod 64 banks)

__global__ __launch_bounds__(512, 1)
void lstm_scan(const __bf16* __restrict__ xswz,
               const __bf16* __restrict__ whh_swz,
               const __bf16* __restrict__ wcomb_swz,
               const float*  __restrict__ bcomb,
               float* __restrict__ h_last)          // [128,256]
{
    __shared__ __attribute__((aligned(16))) __bf16 whh_l[64 * 4 * 32 * 16]; // 256 KB
    __shared__ __attribute__((aligned(16))) __bf16 h_b[2][16 * HBS];        // 16.9 KB
    __shared__ __attribute__((aligned(16))) __bf16 xstage[2][32 * 16];      // 2 KB

    const int tid  = threadIdx.x;
    const int lane = tid & 31;
    const int wave = tid >> 5;                      // 0..15
    const int blk  = blockIdx.x;                    // 0..7
    const int row  = lane & 15;                     // batch row within tile
    const int ah   = lane >> 4;                     // C-layout M-half select

    const unsigned long long xg_base =
        (unsigned long long)(uintptr_t)(xswz + (size_t)blk * TT * 32 * 16);

    // ---- stage W_hh kc4..7 into LDS via async DMA (tile = mt*4 + (kc-4)) ----
    {
        unsigned lds0 = (unsigned)(uintptr_t)whh_l;
        unsigned long long g0 = (unsigned long long)(uintptr_t)whh_swz;
        for (int i = tid; i < 16384; i += 512) {    // 16384 x b128 = 256 KB
            int ldsTile = i >> 6;                   // 0..255
            int within  = i & 63;
            int mt  = ldsTile >> 2;
            int kcl = ldsTile & 3;
            int gT  = mt * 8 + 4 + kcl;
            async_g2l_b128(lds0 + (unsigned)i * 16,
                           g0 + ((unsigned long long)(gT * 64 + within)) * 16ull);
        }
    }
    for (int i = tid; i < 2 * 16 * HBS; i += 512) ((__bf16*)h_b)[i] = (__bf16)0.f;

    // wave 0: async-prefetch x tile for t = 0 into stage buffer 0
    if (wave == 0) {
        unsigned lds = (unsigned)(uintptr_t)&xstage[0][0] + (unsigned)lane * 32u;
        unsigned long long g = xg_base + (unsigned long long)lane * 32ull;
        async_g2l_b128(lds, g);
        async_g2l_b128(lds + 16u, g + 16ull);
    }

    // ---- register-resident tiles: W_hh kc0..3 and W_comb, for mt = 4w..4w+3
    v16bf Bh[4][4];     // [kc][j]
    v16bf Bc[4];        // [j]
    #pragma unroll
    for (int j = 0; j < 4; ++j) {
        int mt = 4 * wave + j;
        Bc[j] = ld32B(wcomb_swz + ((size_t)(mt * 32 + lane)) * 16);
        #pragma unroll
        for (int kc = 0; kc < 4; ++kc)
            Bh[kc][j] = ld32B(whh_swz + ((size_t)((mt * 8 + kc) * 32 + lane)) * 16);
    }

    // ---- packed bf16 biases for this lane's 32 gate columns (stay packed) ----
    unsigned bias_pk[16];
    #pragma unroll
    for (int j = 0; j < 4; ++j) {
        int colbase = (4 * wave + j) * 16 + 8 * ah;
        #pragma unroll
        for (int rp = 0; rp < 4; ++rp) {
            float f0 = bcomb[perm_orig(colbase + 2 * rp)];
            float f1 = bcomb[perm_orig(colbase + 2 * rp + 1)];
            bias_pk[j * 4 + rp] =
                (unsigned)bf16bits(f0) | ((unsigned)bf16bits(f1) << 16);
        }
    }

    float c_reg[8];
    #pragma unroll
    for (int i = 0; i < 8; ++i) c_reg[i] = 0.f;

    wait_async0();          // W_hh LDS staging + x(t=0) prefetch complete
    __syncthreads();

    for (int t = 0; t < TT; ++t) {
        const int rb = t & 1, wb = rb ^ 1;
        const __bf16* hread = h_b[rb];
        __bf16*       hwrt  = h_b[wb];

        // wave 0: kick off async prefetch of x tile for t+1 (other buffer)
        if (wave == 0 && t + 1 < TT) {
            unsigned lds = (unsigned)(uintptr_t)&xstage[(t + 1) & 1][0]
                           + (unsigned)lane * 32u;
            unsigned long long g = xg_base
                + ((unsigned long long)(t + 1) * 32ull + (unsigned long long)lane) * 32ull;
            async_g2l_b128(lds, g);
            async_g2l_b128(lds + 16u, g + 16ull);
        }

        // x_t B-tile from LDS staging (broadcast reads)
        v16bf bx = ld32B(&xstage[rb][lane * 16]);

        // acc = W_comb x x_t  (C = inline 0 -> no registers)
        v8f acc[4];
        #pragma unroll
        for (int j = 0; j < 4; ++j) {
            v8f z = {0.f, 0.f, 0.f, 0.f, 0.f, 0.f, 0.f, 0.f};
            acc[j] = wmma_bf16(Bc[j], bx, z);
        }

        // recurrent part: kc0..3 from registers, kc4..7 from LDS
        #pragma unroll
        for (int kc = 0; kc < 8; ++kc) {
            v16bf bh = ld32B(&hread[row * HBS + kc * 32 + ah * 16]);
            if (kc < 4) {
                #pragma unroll
                for (int j = 0; j < 4; ++j)
                    acc[j] = wmma_bf16(Bh[kc][j], bh, acc[j]);
            } else {
                #pragma unroll
                for (int j = 0; j < 4; ++j) {
                    int mt = 4 * wave + j;
                    v16bf aw = ld32B(&whh_l[((size_t)((mt * 4 + (kc - 4)) * 32 + lane)) * 16]);
                    acc[j] = wmma_bf16(aw, bh, acc[j]);
                }
            }
        }

        // lane-local cell update: acc[j][0..3]+bias = i,f,g,o of chA; [4..7] chA+1
        #pragma unroll
        for (int j = 0; j < 4; ++j) {
            int chA = (4 * wave + j) * 4 + 2 * ah;
            float c0 = c_reg[2 * j], c1 = c_reg[2 * j + 1];

            float iv = acc[j][0] + unpk(bias_pk[j * 4 + 0], 0);
            float fv = acc[j][1] + unpk(bias_pk[j * 4 + 0], 1);
            float gv = acc[j][2] + unpk(bias_pk[j * 4 + 1], 0);
            float ov = acc[j][3] + unpk(bias_pk[j * 4 + 1], 1);
            c0 = sigm(fv) * c0 + sigm(iv) * ftanh(gv);
            float h0 = sigm(ov) * ftanh(c0);

            iv = acc[j][4] + unpk(bias_pk[j * 4 + 2], 0);
            fv = acc[j][5] + unpk(bias_pk[j * 4 + 2], 1);
            gv = acc[j][6] + unpk(bias_pk[j * 4 + 3], 0);
            ov = acc[j][7] + unpk(bias_pk[j * 4 + 3], 1);
            c1 = sigm(fv) * c1 + sigm(iv) * ftanh(gv);
            float h1 = sigm(ov) * ftanh(c1);

            c_reg[2 * j] = c0; c_reg[2 * j + 1] = c1;
            hwrt[row * HBS + chA]     = (__bf16)h0;
            hwrt[row * HBS + chA + 1] = (__bf16)h1;
            if (t == TT - 1) {
                h_last[((size_t)blk * 16 + row) * HH + chA]     = h0;
                h_last[((size_t)blk * 16 + row) * HH + chA + 1] = h1;
            }
        }

        // wave 0 ensures the t+1 x-stage DMA landed before releasing barrier
        if (wave == 0) wait_async0();
        __syncthreads();
    }
}

// ---------------------------------------------------------------------------
// Kernel 6: EMA chunk scan (associative reformulation). grid (128,16) x 256
// ---------------------------------------------------------------------------
__global__ void ema_chunk(const float* __restrict__ x,
                          const float* __restrict__ W_proj, const float* __restrict__ b_proj,
                          const float* __restrict__ W_gate, const float* __restrict__ b_gate,
                          float* __restrict__ Aout, float* __restrict__ Bout)
{
    int b = blockIdx.x, c = blockIdx.y, h = threadIdx.x;
    float wp[32], wg[32];
    const float4* p4 = (const float4*)(W_proj + (size_t)h * DIN);
    const float4* g4 = (const float4*)(W_gate + (size_t)h * DIN);
    #pragma unroll
    for (int q = 0; q < 8; ++q) {
        float4 a = p4[q], bb = g4[q];
        wp[q*4+0]=a.x; wp[q*4+1]=a.y; wp[q*4+2]=a.z; wp[q*4+3]=a.w;
        wg[q*4+0]=bb.x; wg[q*4+1]=bb.y; wg[q*4+2]=bb.z; wg[q*4+3]=bb.w;
    }
    const float bpv = b_proj[h], bgv = b_gate[h];
    const float* xr = x + ((size_t)b * TT + (size_t)c * (TT / NCHUNK)) * DIN;
    float A = 1.f, Bv = 0.f;
    for (int s = 0; s < TT / NCHUNK; ++s) {
        float z = bpv, gl = bgv;
        #pragma unroll
        for (int k = 0; k < 32; ++k) {
            float xv = xr[s * DIN + k];
            z  += xv * wp[k];
            gl += xv * wg[k];
        }
        float g = sigm(gl);
        Bv = g * z + (1.f - g) * Bv;
        A *= (1.f - g);
    }
    size_t o = ((size_t)b * NCHUNK + c) * HH + h;
    Aout[o] = A;
    Bout[o] = Bv;
}

// grid 128 x 256
__global__ void ema_combine(const float* __restrict__ Ain, const float* __restrict__ Bin,
                            float* __restrict__ ema_last)
{
    int b = blockIdx.x, h = threadIdx.x;
    float e = 0.f;
    for (int c = 0; c < NCHUNK; ++c) {
        size_t o = ((size_t)b * NCHUNK + c) * HH + h;
        e = Bin[o] + Ain[o] * e;
    }
    ema_last[(size_t)b * HH + h] = e;
}

// ---------------------------------------------------------------------------
// Kernel 7: concat + LayerNorm + GELU MLP head. grid 128 x 256
// ---------------------------------------------------------------------------
__global__ void head(const float* __restrict__ h_last, const float* __restrict__ ema_last,
                     const float* __restrict__ ln_s, const float* __restrict__ ln_b,
                     const float* __restrict__ W1, const float* __restrict__ b1,
                     const float* __restrict__ W2, const float* __restrict__ b2,
                     float* __restrict__ out)
{
    __shared__ float v[512];
    __shared__ float red[256];
    int b = blockIdx.x, t = threadIdx.x;
    v[t]       = h_last[(size_t)b * HH + t];
    v[256 + t] = ema_last[(size_t)b * HH + t];
    __syncthreads();

    red[t] = v[t] + v[t + 256];
    __syncthreads();
    for (int o = 128; o > 0; o >>= 1) { if (t < o) red[t] += red[t + o]; __syncthreads(); }
    float mu = red[0] * (1.f / 512.f);
    __syncthreads();

    float d0 = v[t] - mu, d1 = v[t + 256] - mu;
    red[t] = d0 * d0 + d1 * d1;
    __syncthreads();
    for (int o = 128; o > 0; o >>= 1) { if (t < o) red[t] += red[t + o]; __syncthreads(); }
    float rstd = rsqrtf(red[0] * (1.f / 512.f) + 1e-5f);
    __syncthreads();

    v[t]       = d0 * rstd * ln_s[t]       + ln_b[t];
    v[256 + t] = d1 * rstd * ln_s[256 + t] + ln_b[256 + t];
    __syncthreads();

    float acc = b1[t];
    for (int k = 0; k < 512; ++k) acc += v[k] * W1[(size_t)t * 512 + k];
    float g = 0.5f * acc * (1.f + erff(acc * 0.70710678118654752f));  // exact GELU

    red[t] = g * W2[t];
    __syncthreads();
    for (int o = 128; o > 0; o >>= 1) { if (t < o) red[t] += red[t + o]; __syncthreads(); }
    if (t == 0) out[b] = red[0] + b2[0];
}

// ---------------------------------------------------------------------------
// host
// ---------------------------------------------------------------------------
extern "C" void kernel_launch(void* const* d_in, const int* in_sizes, int n_in,
                              void* d_out, int out_size, void* d_ws, size_t ws_size,
                              hipStream_t stream)
{
    const float* x      = (const float*)d_in[0];
    const float* W_in   = (const float*)d_in[1];
    const float* b_in   = (const float*)d_in[2];
    const float* W_ih   = (const float*)d_in[3];
    const float* W_hh   = (const float*)d_in[4];
    const float* b_ih   = (const float*)d_in[5];
    const float* b_hh   = (const float*)d_in[6];
    const float* W_proj = (const float*)d_in[7];
    const float* b_proj = (const float*)d_in[8];
    const float* W_gate = (const float*)d_in[9];
    const float* b_gate = (const float*)d_in[10];
    const float* ln_s   = (const float*)d_in[11];
    const float* ln_b   = (const float*)d_in[12];
    const float* W_h1   = (const float*)d_in[13];
    const float* b_h1   = (const float*)d_in[14];
    const float* W_h2   = (const float*)d_in[15];
    const float* b_h2   = (const float*)d_in[16];

    char* ws = (char*)d_ws;
    float*  wcomb     = (float*)(ws + OFF_WCOMB_F32);
    float*  bcomb     = (float*)(ws + OFF_BCOMB);
    __bf16* whh_swz   = (__bf16*)(ws + OFF_WHH_SWZ);
    __bf16* wcomb_swz = (__bf16*)(ws + OFF_WCOMB_SWZ);
    float*  h_last    = (float*)(ws + OFF_HLAST);
    float*  ema_last  = (float*)(ws + OFF_EMALAST);
    float*  emaA      = (float*)(ws + OFF_EMA_A);
    float*  emaB      = (float*)(ws + OFF_EMA_B);
    __bf16* xswz      = (__bf16*)(ws + OFF_XSWZ);

    prep_wcomb<<<G4, 32, 0, stream>>>(W_ih, W_in, b_in, b_ih, b_hh, wcomb, bcomb);
    pack_whh<<<512, 256, 0, stream>>>(W_hh, whh_swz);
    pack_wcomb<<<64, 256, 0, stream>>>(wcomb, wcomb_swz);
    pack_x<<<2048, 256, 0, stream>>>(x, xswz);

    lstm_scan<<<NWG, 512, 0, stream>>>(xswz, whh_swz, wcomb_swz, bcomb, h_last);

    ema_chunk<<<dim3(BB, NCHUNK), HH, 0, stream>>>(x, W_proj, b_proj, W_gate, b_gate,
                                                   emaA, emaB);
    ema_combine<<<BB, HH, 0, stream>>>(emaA, emaB, ema_last);

    head<<<BB, HH, 0, stream>>>(h_last, ema_last, ln_s, ln_b,
                                W_h1, b_h1, W_h2, b_h2, (float*)d_out);
}
